// ReparameterizedGMM_Linear_41781441856011
// MI455X (gfx1250) — matrix operand order repaired and checked by reading, not compiled, not run on hardware
//
#include <hip/hip_runtime.h>
#include <hip/hip_bf16.h>

#define D_DIM 2048
#define K_EXP 10
#define BS_N  1024

typedef __attribute__((ext_vector_type(16))) __bf16 v16bf;
typedef __attribute__((ext_vector_type(8)))  __bf16 v8bf;
typedef __attribute__((ext_vector_type(4)))  __bf16 v4bf;
typedef __attribute__((ext_vector_type(8)))  float  v8f;

// ---------------------------------------------------------------------------
// Kernel 1: counting-sort samples by expert (offsets[K+1], order[BS] in ws).
// Output is independent of intra-expert ordering, so atomics are safe.
// ---------------------------------------------------------------------------
__global__ __launch_bounds__(BS_N) void build_order_kernel(
    const int* __restrict__ comp, int* __restrict__ offsets,
    int* __restrict__ order) {
  __shared__ int cnt[K_EXP];
  __shared__ int cur[K_EXP];
  const int t = threadIdx.x;
  if (t < K_EXP) cnt[t] = 0;
  __syncthreads();
  const int c = comp[t];
  atomicAdd(&cnt[c], 1);
  __syncthreads();
  if (t == 0) {
    int acc = 0;
    for (int k = 0; k < K_EXP; ++k) {
      offsets[k] = acc;
      cur[k] = acc;
      acc += cnt[k];
    }
    offsets[K_EXP] = acc;
  }
  __syncthreads();
  const int pos = atomicAdd(&cur[c], 1);
  order[pos] = t;
}

// ---------------------------------------------------------------------------
// Kernel 2: one-time fp32 -> (bf16 hi, bf16 lo) split, 4 elems/thread.
// Removes the recurring conversion VALU from the GEMM inner loop.
// ---------------------------------------------------------------------------
__global__ __launch_bounds__(256) void split_kernel(
    const float* __restrict__ in, __bf16* __restrict__ hi,
    __bf16* __restrict__ lo, long long n4) {
  const long long i = (long long)blockIdx.x * blockDim.x + threadIdx.x;
  if (i >= n4) return;
  const float4 f = ((const float4*)in)[i];
  v4bf h, l;
  float fv[4] = {f.x, f.y, f.z, f.w};
#pragma unroll
  for (int j = 0; j < 4; ++j) {
    __bf16 hh = (__bf16)fv[j];
    h[j] = hh;
    l[j] = (__bf16)(fv[j] - (float)hh);
  }
  *(v4bf*)(hi + 4 * i) = h;
  *(v4bf*)(lo + 4 * i) = l;
}

// Combine two contiguous 16B bf16 chunks into a WMMA fragment.
__device__ __forceinline__ v16bf ldfrag2(const __bf16* p0, const __bf16* p1) {
  v8bf a = *(const v8bf*)p0;
  v8bf b = *(const v8bf*)p1;
  return __builtin_shufflevector(a, b, 0, 1, 2, 3, 4, 5, 6, 7, 8, 9, 10, 11,
                                 12, 13, 14, 15);
}

// Inline fallback: load 2x8 fp32 and split to hi/lo bf16 fragments.
__device__ __forceinline__ void ldfrag_split(const float* p0, const float* p1,
                                             v16bf& h, v16bf& l) {
  float av[16] __attribute__((aligned(16)));
  *(float4*)(av + 0)  = *(const float4*)(p0);
  *(float4*)(av + 4)  = *(const float4*)(p0 + 4);
  *(float4*)(av + 8)  = *(const float4*)(p1);
  *(float4*)(av + 12) = *(const float4*)(p1 + 4);
#pragma unroll
  for (int i = 0; i < 16; ++i) {
    __bf16 hh = (__bf16)av[i];
    h[i] = hh;
    l[i] = (__bf16)(av[i] - (float)hh);
  }
}

// ---------------------------------------------------------------------------
// Kernel 3: grouped GEMM. One wave32 per (expert k, 16-sample tile,
// 32-column e-tile = two 16x16 N-subtiles). bf16x3 split product
// (hi*hi + hi*lo + lo*hi, fp32 accum) restores ~fp32 accuracy.
// PA/PB: operand pre-split in workspace vs inline conversion.
// ---------------------------------------------------------------------------
template <bool PA, bool PB>
__global__ __launch_bounds__(256) void gmm_wmma_kernel(
    const float* __restrict__ z, const float* __restrict__ mvec,
    const float* __restrict__ L, const __bf16* __restrict__ zh,
    const __bf16* __restrict__ zl, const __bf16* __restrict__ Lh,
    const __bf16* __restrict__ Ll, const int* __restrict__ offsets,
    const int* __restrict__ order, float* __restrict__ out) {
  const int wave = threadIdx.x >> 5;  // 8 waves per block
  const int lane = threadIdx.x & 31;
  const int task = blockIdx.x * 8 + wave;

  const int etile = task & 63;        // 64 e-pairs (D / 32)
  const int rest  = task >> 6;
  const int k     = rest % K_EXP;
  const int stile = rest / K_EXP;     // worst case 64 sample tiles

  const int start = offsets[k];
  const int end   = offsets[k + 1];
  const int row0  = start + stile * 16;
  if (row0 >= end) return;            // wave-uniform -> EXEC stays all-ones

  const int lh = lane & 15;
  const int hi = lane >> 4;

  // A-matrix row (M = lh for both lane halves); clamp partial tiles.
  int arow = row0 + lh;
  if (arow >= end) arow = end - 1;
  const size_t asrc = (size_t)order[arow] * D_DIM;

  const int e0 = etile * 32;
  const size_t b0src = ((size_t)k * D_DIM + (size_t)(e0 + lh)) * D_DIM;
  const size_t b1src = b0src + (size_t)16 * D_DIM;

  v8f acc00 = {}, acc01 = {}, acc02 = {};  // N-subtile 0
  v8f acc10 = {}, acc11 = {}, acc12 = {};  // N-subtile 1

  for (int d0 = 0; d0 < D_DIM; d0 += 32) {
    v16bf ah, al, bh0, bl0, bh1, bl1;

    // A fragment: 16-bit A 16x32 layout -> two contiguous 8-elem chunks.
    const int aoff0 = d0 + hi * 8;
    const int aoff1 = d0 + 16 + hi * 8;
    if constexpr (PA) {
      ah = ldfrag2(zh + asrc + aoff0, zh + asrc + aoff1);
      al = ldfrag2(zl + asrc + aoff0, zl + asrc + aoff1);
    } else {
      ldfrag_split(z + asrc + aoff0, z + asrc + aoff1, ah, al);
    }

    // B fragments: 16-bit B 32x16 layout -> two contiguous 8-elem chunks of an L row.
    const int boff = d0 + hi * 8;
    if constexpr (PB) {
      bh0 = ldfrag2(Lh + b0src + boff, Lh + b0src + boff + 16);
      bl0 = ldfrag2(Ll + b0src + boff, Ll + b0src + boff + 16);
      bh1 = ldfrag2(Lh + b1src + boff, Lh + b1src + boff + 16);
      bl1 = ldfrag2(Ll + b1src + boff, Ll + b1src + boff + 16);
    } else {
      ldfrag_split(L + b0src + boff, L + b0src + boff + 16, bh0, bl0);
      ldfrag_split(L + b1src + boff, L + b1src + boff + 16, bh1, bl1);
    }

    // Six independent accumulation chains on the matrix pipe.
    acc00 = __builtin_amdgcn_wmma_f32_16x16x32_bf16(
        false, ah, false, bh0, (short)0, acc00, false, false);
    acc01 = __builtin_amdgcn_wmma_f32_16x16x32_bf16(
        false, ah, false, bl0, (short)0, acc01, false, false);
    acc02 = __builtin_amdgcn_wmma_f32_16x16x32_bf16(
        false, al, false, bh0, (short)0, acc02, false, false);
    acc10 = __builtin_amdgcn_wmma_f32_16x16x32_bf16(
        false, ah, false, bh1, (short)0, acc10, false, false);
    acc11 = __builtin_amdgcn_wmma_f32_16x16x32_bf16(
        false, ah, false, bl1, (short)0, acc11, false, false);
    acc12 = __builtin_amdgcn_wmma_f32_16x16x32_bf16(
        false, al, false, bh1, (short)0, acc12, false, false);
  }

  // Epilogue: C/D layout -> VGPR j: lanes 0-15 M=j, lanes 16-31 M=j+8,
  // N = lane&15. Add m[k,e], guard partial tiles, scatter by sample id.
  const float mb0 = mvec[(size_t)k * D_DIM + e0 + lh];
  const float mb1 = mvec[(size_t)k * D_DIM + e0 + 16 + lh];
#pragma unroll
  for (int j = 0; j < 8; ++j) {
    const int r = row0 + j + hi * 8;
    if (r < end) {
      float* orow = out + (size_t)order[r] * D_DIM;
      orow[e0 + lh]      = acc00[j] + acc01[j] + acc02[j] + mb0;
      orow[e0 + 16 + lh] = acc10[j] + acc11[j] + acc12[j] + mb1;
    }
  }
}

extern "C" void kernel_launch(void* const* d_in, const int* in_sizes, int n_in,
                              void* d_out, int out_size, void* d_ws,
                              size_t ws_size, hipStream_t stream) {
  const float* z    = (const float*)d_in[0];  // (BS, D)
  const float* m    = (const float*)d_in[1];  // (K, D)
  const float* L    = (const float*)d_in[2];  // (K, D, D)
  const int*   comp = (const int*)d_in[3];    // (BS,)
  float* out = (float*)d_out;                 // (BS, D)

  const size_t zN = (size_t)BS_N * D_DIM;           // 2,097,152
  const size_t LN = (size_t)K_EXP * D_DIM * D_DIM;  // 41,943,040

  char* base   = (char*)d_ws;
  int* offsets = (int*)base;                  // K+1 ints
  int* order   = (int*)(base + 256);          // BS ints
  char* payload = base + 16 * 1024;
  __bf16* zh = (__bf16*)payload;
  __bf16* zl = zh + zN;
  __bf16* Lh = zl + zN;
  __bf16* Ll = Lh + LN;

  const size_t need_z    = 16 * 1024 + 2 * zN * sizeof(__bf16);
  const size_t need_full = need_z + 2 * LN * sizeof(__bf16);

  build_order_kernel<<<1, BS_N, 0, stream>>>(comp, offsets, order);

  // Worst-case: 64 sample-tiles x K experts x 64 e-pairs = 40960 wave tasks
  // = 5120 blocks of 8 waves; inactive waves exit on one scalar compare.
  const int blocks = (64 * K_EXP * 64) / 8;

  if (ws_size >= need_full) {
    split_kernel<<<(int)((zN / 4 + 255) / 256), 256, 0, stream>>>(
        z, zh, zl, (long long)(zN / 4));
    split_kernel<<<(int)((LN / 4 + 255) / 256), 256, 0, stream>>>(
        L, Lh, Ll, (long long)(LN / 4));
    gmm_wmma_kernel<true, true><<<blocks, 256, 0, stream>>>(
        z, m, L, zh, zl, Lh, Ll, offsets, order, out);
  } else if (ws_size >= need_z) {
    split_kernel<<<(int)((zN / 4 + 255) / 256), 256, 0, stream>>>(
        z, zh, zl, (long long)(zN / 4));
    gmm_wmma_kernel<true, false><<<blocks, 256, 0, stream>>>(
        z, m, L, zh, zl, nullptr, nullptr, offsets, order, out);
  } else {
    gmm_wmma_kernel<false, false><<<blocks, 256, 0, stream>>>(
        z, m, L, nullptr, nullptr, nullptr, nullptr, offsets, order, out);
  }
}
